// MultiHeadedAttention_1932735283883
// MI455X (gfx1250) — matrix operand order repaired and checked by reading, not compiled, not run on hardware
//
#include <hip/hip_runtime.h>

// ---------------------------------------------------------------------------
// MHA forward for MI455X (gfx1250, wave32, WMMA bf16).
// B=2, S=2048, D=1024, H=16, dh=64.
// Pipeline: projections write fp32 outputs (present) AND bf16 shadow copies
// for the attention path; attention consumes/produces bf16 (async LDS staging
// via global_load_async_to_lds_b128 / s_wait_asynccnt); final GEMM reads
// bf16 A directly.  All matrix math on v_wmma_f32_16x16x32_bf16.
// ---------------------------------------------------------------------------

#define B_  2
#define S_  2048
#define D_  1024
#define H_  16
#define DH_ 64
#define NEL_ ((size_t)B_ * S_ * D_)   // 4,194,304 elements

typedef __attribute__((ext_vector_type(16))) __bf16 v16bf;
typedef __attribute__((ext_vector_type(8)))  float  v8f;
typedef __attribute__((ext_vector_type(4)))  float  f4;
typedef __attribute__((ext_vector_type(4)))  unsigned int u4;
typedef __attribute__((ext_vector_type(4)))  int    i4;

#define AS1 __attribute__((address_space(1)))
#define AS3 __attribute__((address_space(3)))

__device__ __forceinline__ __bf16 f2bf(float x) { return (__bf16)x; }

#if defined(__has_builtin)
#if __has_builtin(__builtin_amdgcn_global_load_async_to_lds_b128)
#define HAVE_ASYNC 1
#endif
#endif
#ifndef HAVE_ASYNC
#define HAVE_ASYNC 0
#endif

__device__ __forceinline__ void wait_async0() {
#if defined(__has_builtin) && __has_builtin(__builtin_amdgcn_s_wait_asynccnt)
    __builtin_amdgcn_s_wait_asynccnt(0);
#else
    asm volatile("s_wait_asynccnt 0" ::: "memory");
#endif
}

// Copy 16 bytes global -> LDS. Async DMA (ASYNCcnt) when available,
// else a plain b128 load/ds_store pair.
__device__ __forceinline__ void cp16_g2l(const void* g, void* l) {
#if HAVE_ASYNC
    __builtin_amdgcn_global_load_async_to_lds_b128(
        (AS1 i4*)(uintptr_t)g,   // global src: int4 ptr in device (AS1) space
        (AS3 i4*)l,              // LDS dst:    int4 ptr in local  (AS3) space
        0, 0);
#else
    *(u4*)l = *(const u4*)g;
#endif
}

// ===========================================================================
// GEMM:  C[M,N] = A[M,K] @ W[N,K]^T + bias[N]      (M=4096, N=K=1024)
//   Block tile 128(M) x 64(N), 256 threads = 8 waves; wave w owns a 16x64
//   strip (4 f32 accumulators), K-loop in 32-wide slices through LDS.
//   AIN: 0 = A is fp32 (convert at staging), 1 = A is bf16 (async-copy tiles)
//   OM : 0 = fp32 [M,N]
//        1 = fp32 `present` [B,2,H,S,dh] (kv folded into outf) + bf16 shadow
//            copy [B,H,S,dh] (kv folded into outb)
//        2 = bf16 [M,N]
// ===========================================================================
template <int AIN, int OM>
__global__ __launch_bounds__(256) void gemm_bt(
    const void* __restrict__ Ain, const float* __restrict__ W,
    const float* __restrict__ bias, float* __restrict__ outf,
    __bf16* __restrict__ outb)
{
    __shared__ __bf16 sA[128][40];   // 32-wide K slice + pad
    __shared__ __bf16 sW[64][40];

    const int t    = threadIdx.x;
    const int wv   = t >> 5;
    const int lane = t & 31;
    const int lm   = lane & 15;
    const int half = lane >> 4;
    const int n0   = blockIdx.x * 64;
    const int m0   = blockIdx.y * 128;
    const int K    = D_;
    const int N    = D_;

    v8f acc[4] = {};

    const int arow = t >> 1, acol = (t & 1) * 16;   // 128 rows x 32 cols
    const int wrow = t >> 2, wcol = (t & 3) * 8;    // 64 rows x 32 cols

    for (int k0 = 0; k0 < K; k0 += 32) {
        // ---- stage A tile --------------------------------------------------
        if (AIN == 0) {
            const float* p = (const float*)Ain + (size_t)(m0 + arow) * K + k0 + acol;
            if (k0 + 32 < K) __builtin_prefetch(p + 32, 0, 0);   // global_prefetch_b8
#pragma unroll
            for (int i = 0; i < 4; ++i) {
                f4 v = *(const f4*)(p + 4 * i);
#pragma unroll
                for (int j = 0; j < 4; ++j) sA[arow][acol + 4 * i + j] = f2bf(v[j]);
            }
        } else {
            const __bf16* p = (const __bf16*)Ain + (size_t)(m0 + arow) * K + k0 + acol;
            char* l = (char*)&sA[arow][acol];
            cp16_g2l(p, l);
            cp16_g2l((const char*)p + 16, l + 16);
        }
        // ---- stage W tile (always fp32 -> bf16) ----------------------------
        {
            const float* p = W + (size_t)(n0 + wrow) * K + k0 + wcol;
            if (k0 + 32 < K) __builtin_prefetch(p + 32, 0, 0);
#pragma unroll
            for (int i = 0; i < 2; ++i) {
                f4 v = *(const f4*)(p + 4 * i);
#pragma unroll
                for (int j = 0; j < 4; ++j) sW[wrow][wcol + 4 * i + j] = f2bf(v[j]);
            }
        }
        if (AIN == 1) wait_async0();
        __syncthreads();

        // A fragment (16x32, contiguous K = half*16 + 2v + p): lane = M, K split by half.
        v16bf afrag;
#pragma unroll
        for (int vi = 0; vi < 8; ++vi)
#pragma unroll
            for (int p = 0; p < 2; ++p) {
                const int kk = half * 16 + vi * 2 + p;
                afrag[2 * vi + p] = sA[(wv << 4) + lm][kk];
            }
#pragma unroll
        for (int nt = 0; nt < 4; ++nt) {
            // B fragment (32x16): lane = N, K = half*16 + 2v + p ; B[k][n] = W[n][k]
            v16bf bfrag;
#pragma unroll
            for (int vi = 0; vi < 8; ++vi)
#pragma unroll
                for (int p = 0; p < 2; ++p)
                    bfrag[2 * vi + p] = sW[nt * 16 + lm][half * 16 + 2 * vi + p];

            acc[nt] = __builtin_amdgcn_wmma_f32_16x16x32_bf16(
                false, afrag, false, bfrag, (short)0, acc[nt], false, false);
        }
        __syncthreads();
    }

    // ---- epilogue: C layout M = r + 8*half, N = lane%16 --------------------
#pragma unroll
    for (int nt = 0; nt < 4; ++nt) {
        const int   gn = n0 + nt * 16 + lm;
        const float bv = bias[gn];
#pragma unroll
        for (int r = 0; r < 8; ++r) {
            const int   gm  = m0 + (wv << 4) + r + 8 * half;
            const float val = acc[nt][r] + bv;
            if (OM == 0) {
                outf[(size_t)gm * N + gn] = val;
            } else if (OM == 2) {
                outb[(size_t)gm * N + gn] = f2bf(val);
            } else {
                const int b = gm >> 11, s = gm & (S_ - 1);
                const int h = gn >> 6,  d = gn & (DH_ - 1);
                outf[(size_t)b * (2 * H_ * S_ * DH_) + (size_t)h * (S_ * DH_) +
                     (size_t)s * DH_ + d] = val;                       // present (fp32)
                outb[(size_t)b * (H_ * S_ * DH_) + (size_t)h * (S_ * DH_) +
                     (size_t)s * DH_ + d] = f2bf(val);                 // bf16 shadow
            }
        }
    }
}

// ===========================================================================
// Flash attention per (b, h):  O = softmax(Q K^T / 8 + mask) V
//   Block = 4 waves; wave w owns Q rows [q0, q0+16).  Q/K/V consumed as bf16
//   (produced by the projection epilogues); K/V tiles staged via async
//   global->LDS copies.  Output written bf16 for the final GEMM.
// ===========================================================================
__global__ __launch_bounds__(128) void attn_kernel(
    const __bf16* __restrict__ Qb,   // [B,S,D] bf16
    const __bf16* __restrict__ Kb,   // [B,H,S,DH] bf16
    const __bf16* __restrict__ Vb,   // [B,H,S,DH] bf16
    const float*  __restrict__ mask, // [B,1,S,S] fp32
    __bf16* __restrict__ O)          // [B,S,D] bf16
{
    __shared__ __bf16 sK[64][72];
    __shared__ __bf16 sV[64][72];
    __shared__ __bf16 sP[4][16][72];

    const int t    = threadIdx.x;
    const int wv   = t >> 5;
    const int lane = t & 31;
    const int lm   = lane & 15;
    const int half = lane >> 4;
    const int bh   = blockIdx.y;
    const int b    = bh >> 4, h = bh & 15;
    const int q0   = blockIdx.x * 64 + wv * 16;
    const float scale = 0.125f;      // 1/sqrt(64)

    const char* Kslab = (const char*)(Kb + ((size_t)b * H_ + h) * (size_t)(S_ * DH_));
    const char* Vslab = (const char*)(Vb + ((size_t)b * H_ + h) * (size_t)(S_ * DH_));

    // Q fragments live in registers for the whole key loop (dh=64 -> 2 frags).
    v16bf aq[2];
#pragma unroll
    for (int ks = 0; ks < 2; ++ks)
#pragma unroll
        for (int vi = 0; vi < 8; ++vi)
#pragma unroll
            for (int p = 0; p < 2; ++p) {
                const int kk = half * 16 + vi * 2 + p;
                aq[ks][2 * vi + p] =
                    Qb[((size_t)b * S_ + q0 + lm) * D_ + h * DH_ + ks * 32 + kk];
            }

    v8f   accO[4] = {};
    float mrun[8], lrun[8];
#pragma unroll
    for (int r = 0; r < 8; ++r) { mrun[r] = -1e30f; lrun[r] = 0.0f; }

    for (int kt = 0; kt < S_; kt += 64) {
        __syncthreads();   // previous iteration's reads done before overwrite
        {   // stage 64x64 bf16 K,V tiles: 8KB each, contiguous in global.
            // 128 threads x 4 chunks x 16B; LDS rows padded to 144B.
#pragma unroll
            for (int i = 0; i < 4; ++i) {
                const int off = t * 16 + i * 2048;      // byte offset in tile
                const int row = off >> 7, wb = off & 127;
                cp16_g2l(Kslab + (size_t)kt * 128 + off,
                         (char*)&sK[0][0] + row * 144 + wb);
                cp16_g2l(Vslab + (size_t)kt * 128 + off,
                         (char*)&sV[0][0] + row * 144 + wb);
            }
        }
        wait_async0();
        __syncthreads();

        // ---- scores: S[16 x 64keys] = Q K^T (4 n-tiles x 2 k-steps) --------
        v8f sc[4] = {};
#pragma unroll
        for (int nt = 0; nt < 4; ++nt)
#pragma unroll
            for (int ks = 0; ks < 2; ++ks) {
                v16bf bk;   // B[k=dh][n=key] = K[key][dh]
#pragma unroll
                for (int vi = 0; vi < 8; ++vi)
#pragma unroll
                    for (int p = 0; p < 2; ++p)
                        bk[2 * vi + p] =
                            sK[nt * 16 + lm][ks * 32 + half * 16 + 2 * vi + p];
                sc[nt] = __builtin_amdgcn_wmma_f32_16x16x32_bf16(
                    false, aq[ks], false, bk, (short)0, sc[nt], false, false);
            }

        // ---- scale + mask, row max -----------------------------------------
        float rowmax[8];
#pragma unroll
        for (int r = 0; r < 8; ++r) rowmax[r] = -1e30f;
#pragma unroll
        for (int nt = 0; nt < 4; ++nt)
#pragma unroll
            for (int r = 0; r < 8; ++r) {
                const float mval = mask[(size_t)b * S_ * S_ +
                                        (size_t)(q0 + r + 8 * half) * S_ +
                                        (kt + nt * 16 + lm)];
                const float sval = sc[nt][r] * scale + mval;
                sc[nt][r] = sval;
                rowmax[r] = fmaxf(rowmax[r], sval);
            }
#pragma unroll
        for (int r = 0; r < 8; ++r) {   // reduce over 16 columns (stays in half)
            rowmax[r] = fmaxf(rowmax[r], __shfl_xor(rowmax[r], 1, 32));
            rowmax[r] = fmaxf(rowmax[r], __shfl_xor(rowmax[r], 2, 32));
            rowmax[r] = fmaxf(rowmax[r], __shfl_xor(rowmax[r], 4, 32));
            rowmax[r] = fmaxf(rowmax[r], __shfl_xor(rowmax[r], 8, 32));
        }

        // ---- online softmax update -----------------------------------------
#pragma unroll
        for (int r = 0; r < 8; ++r) {
            const float mnew  = fmaxf(mrun[r], rowmax[r]);
            const float alpha = __expf(mrun[r] - mnew);
            mrun[r] = mnew;
            float ps = 0.0f;
#pragma unroll
            for (int nt = 0; nt < 4; ++nt) {
                const float pv = __expf(sc[nt][r] - mnew);
                sc[nt][r] = pv;
                ps += pv;
            }
            ps += __shfl_xor(ps, 1, 32);
            ps += __shfl_xor(ps, 2, 32);
            ps += __shfl_xor(ps, 4, 32);
            ps += __shfl_xor(ps, 8, 32);
            lrun[r] = lrun[r] * alpha + ps;
#pragma unroll
            for (int dt = 0; dt < 4; ++dt) accO[dt][r] *= alpha;
        }

        // ---- P: C-layout -> LDS -> A-layout --------------------------------
#pragma unroll
        for (int nt = 0; nt < 4; ++nt)
#pragma unroll
            for (int r = 0; r < 8; ++r)
                sP[wv][r + 8 * half][nt * 16 + lm] = f2bf(sc[nt][r]);
        asm volatile("s_wait_dscnt 0" ::: "memory");  // same-wave LDS RAW fence

        // ---- O += P[16 x 64keys] @ V[64keys x 64dh] ------------------------
#pragma unroll
        for (int ks = 0; ks < 2; ++ks) {
            v16bf ap;   // A-layout over key dim
#pragma unroll
            for (int vi = 0; vi < 8; ++vi)
#pragma unroll
                for (int p = 0; p < 2; ++p) {
                    const int kk = half * 16 + vi * 2 + p;
                    ap[2 * vi + p] = sP[wv][lm][ks * 32 + kk];
                }
#pragma unroll
            for (int dt = 0; dt < 4; ++dt) {
                v16bf bv;  // B[k=key][n=dh] = V[key][dh]
#pragma unroll
                for (int vi = 0; vi < 8; ++vi)
#pragma unroll
                    for (int p = 0; p < 2; ++p)
                        bv[2 * vi + p] =
                            sV[ks * 32 + half * 16 + 2 * vi + p][dt * 16 + lm];
                accO[dt] = __builtin_amdgcn_wmma_f32_16x16x32_bf16(
                    false, ap, false, bv, (short)0, accO[dt], false, false);
            }
        }
    }

    // ---- normalize + store bf16 [B,S,D] ------------------------------------
#pragma unroll
    for (int dt = 0; dt < 4; ++dt)
#pragma unroll
        for (int r = 0; r < 8; ++r) {
            const float val = accO[dt][r] / lrun[r];
            O[((size_t)b * S_ + q0 + r + 8 * half) * D_ + h * DH_ + dt * 16 + lm] =
                f2bf(val);
        }
}

// ===========================================================================
extern "C" void kernel_launch(void* const* d_in, const int* in_sizes, int n_in,
                              void* d_out, int out_size, void* d_ws, size_t ws_size,
                              hipStream_t stream) {
    (void)in_sizes; (void)n_in; (void)out_size; (void)ws_size;

    const float* key   = (const float*)d_in[0];
    const float* value = (const float*)d_in[1];
    const float* query = (const float*)d_in[2];
    const float* mask  = (const float*)d_in[3];
    const float* Wq = (const float*)d_in[4];  const float* bq = (const float*)d_in[5];
    const float* Wk = (const float*)d_in[6];  const float* bk = (const float*)d_in[7];
    const float* Wv = (const float*)d_in[8];  const float* bv = (const float*)d_in[9];
    const float* Wo = (const float*)d_in[10]; const float* bo = (const float*)d_in[11];

    float* out     = (float*)d_out;
    float* present = out + NEL_;                          // [B,2,H,S,dh] fp32

    __bf16* wsq   = (__bf16*)d_ws;                        // Q   bf16 [B,S,D]    8MB
    __bf16* wsk   = wsq + NEL_;                           // K   bf16 [B,H,S,dh] 8MB
    __bf16* wsv   = wsk + NEL_;                           // V   bf16 [B,H,S,dh] 8MB
    __bf16* attnb = wsv + NEL_;                           // attn bf16 [B,S,D]   8MB

    const dim3 gblk(256);
    const dim3 ggrid(D_ / 64, (B_ * S_) / 128);

    // Projections. K/V dual-write: fp32 present (output) + bf16 shadow for attention.
    gemm_bt<0, 2><<<ggrid, gblk, 0, stream>>>(query, Wq, bq, nullptr, wsq);
    gemm_bt<0, 1><<<ggrid, gblk, 0, stream>>>(key,   Wk, bk, present, wsk);
    gemm_bt<0, 1><<<ggrid, gblk, 0, stream>>>(value, Wv, bv,
                                              present + (size_t)H_ * S_ * DH_, wsv);

    // Flash attention over (b,h) x q-tiles (pure bf16 data path).
    attn_kernel<<<dim3(S_ / 64, B_ * H_), dim3(128), 0, stream>>>(
        wsq, wsk, wsv, mask, attnb);

    // Output projection (bf16 A, async-staged).
    gemm_bt<1, 0><<<ggrid, gblk, 0, stream>>>(attnb, Wo, bo, out, nullptr);
}
